// MultiHeadAttention_71313636983240
// MI455X (gfx1250) — matrix operand order repaired and checked by reading, not compiled, not run on hardware
//
#include <hip/hip_runtime.h>

// ---------------------------------------------------------------------------
// CDNA5 (gfx1250, wave32) fused multi-head attention
//   B=4, L=2048, D=512, H=8, d_k=64.
//   bf16 WMMA (v_wmma_f32_16x16x32_bf16) with f32 accumulation,
//   TDM (tensor_load_to_lds) double-buffered K/V staging in attention.
// ---------------------------------------------------------------------------

typedef __attribute__((ext_vector_type(16))) __bf16        v16bf;
typedef __attribute__((ext_vector_type(8)))  float         v8f;
typedef __attribute__((ext_vector_type(4)))  unsigned int  v4u;
typedef __attribute__((ext_vector_type(8)))  int           v8i;
typedef __attribute__((ext_vector_type(4)))  int           v4i;

union Frag16 {                 // 16 bf16 = 32 bytes = 8 VGPRs (wave32 fragment)
    v16bf          v;
    uint4          q[2];
    unsigned short s[16];
};

__device__ __forceinline__ unsigned short f2bf(float f) {
    unsigned int u = __float_as_uint(f);
    u += 0x7fffu + ((u >> 16) & 1u);          // round-to-nearest-even
    return (unsigned short)(u >> 16);
}

__device__ __forceinline__ v8f wmma_bf16(const Frag16& a, const Frag16& b, v8f c) {
    // D = A(16x32 bf16) * B(32x16 bf16) + C(16x16 f32)
    return __builtin_amdgcn_wmma_f32_16x16x32_bf16(
        false, a.v, false, b.v, (short)0, c, false, false);
}

// ---------------------------------------------------------------------------
// Tensor Data Mover: async 2D tile (global -> LDS), bf16 elements.
// D# built per CDNA5 ISA (group0 + group1; 2D => higher
// groups zero). tensor dims == tile dims (tile fully in-bounds),
// dim0 stride = row pitch.  6-arg builtin form (probe: "expected 6").
// ---------------------------------------------------------------------------
__device__ __forceinline__ void tdm_load_2d(unsigned ldsoff, const void* gptr,
                                            unsigned tile_d0, unsigned tile_d1,
                                            unsigned stride0) {
    unsigned long long ga = (unsigned long long)(size_t)gptr;
    v4u g0 = { 1u,                                   // count=1, user descriptor
               ldsoff,                               // lds_addr (bytes)
               (unsigned)ga,                         // global_addr[31:0]
               (unsigned)((ga >> 32) & 0x01ffffffu)  // global_addr[56:32]
                   | 0x80000000u };                  // type=2 ("image")
    v8i g1 = { (int)0x00010000u,                     // data_size=1 (2 bytes)
               (int)(tile_d0 << 16),                 // tensor_dim0[15:0]
               (int)(tile_d1 << 16),                 // tensor_dim1[15:0]
               (int)(tile_d0 << 16),                 // tile_dim0
               (int)tile_d1,                         // tile_dim1 (tile_dim2=0)
               (int)stride0,                         // tensor_dim0_stride[31:0]
               0, 0 };
    v4i gz4 = { 0, 0, 0, 0 };
    v8i gz8 = { 0, 0, 0, 0, 0, 0, 0, 0 };
    __builtin_amdgcn_tensor_load_to_lds(g0, g1, gz4, gz4, gz8, 0);
}

__device__ __forceinline__ unsigned lds_off(const void* p) {
    // flat LDS aperture: addr[31:0] is the LDS byte offset
    return (unsigned)(size_t)p;
}

// ---------------------------------------------------------------------------
// fp32 -> bf16 bulk convert (8 elems/thread), used for query/key/value.
// ---------------------------------------------------------------------------
__global__ void __launch_bounds__(256) x_conv(const float* __restrict__ X,
                                              unsigned short* __restrict__ Xb) {
    size_t i = ((size_t)blockIdx.x * 256 + threadIdx.x) * 8;
    float4 f0 = *(const float4*)(X + i);
    float4 f1 = *(const float4*)(X + i + 4);
    union { uint4 q; unsigned short s[8]; } o;
    o.s[0] = f2bf(f0.x); o.s[1] = f2bf(f0.y); o.s[2] = f2bf(f0.z); o.s[3] = f2bf(f0.w);
    o.s[4] = f2bf(f1.x); o.s[5] = f2bf(f1.y); o.s[6] = f2bf(f1.z); o.s[7] = f2bf(f1.w);
    *(uint4*)(Xb + i) = o.q;
}

// ---------------------------------------------------------------------------
// Weight transpose + convert: W[k][n] (512x512 f32) -> Wt[n][k] bf16.
// ---------------------------------------------------------------------------
__global__ void __launch_bounds__(256) wt_conv(const float* __restrict__ W,
                                               unsigned short* __restrict__ Wt) {
    int idx = blockIdx.x * 256 + threadIdx.x;        // 262144
    int k = idx >> 9;
    int n = idx & 511;
    Wt[(size_t)n * 512 + k] = f2bf(W[(size_t)k * 512 + n]);
}

// ---------------------------------------------------------------------------
// GEMM: out = Xb(8192x512 bf16) @ W(512x512 pre-transposed bf16) + bias.
// One 16(M) x 64(N) tile per wave: A-fragment reused across 4 WMMAs per K-step.
//   mode 0: q-proj -> bf16 [B,H,L,dk] (1/sqrt(dk) folded)     mode 1: k-proj
//   mode 2: v-proj -> bf16 [B,H,dk,L] (transposed)            mode 3: -> f32 out
// ---------------------------------------------------------------------------
__global__ void __launch_bounds__(256) gemm_proj(const unsigned short* __restrict__ Xb,
                                                 const unsigned short* __restrict__ Wt,
                                                 const float* __restrict__ bias,
                                                 void* __restrict__ outp,
                                                 int mode, float scale) {
    const int wave  = threadIdx.x >> 5;
    const int lane  = threadIdx.x & 31;
    const int l16   = lane & 15;
    const int half  = lane >> 4;
    const int tile  = blockIdx.x * 8 + wave;         // 4096 wave-tiles
    const int mtile = tile & 511;                    // M = 8192 -> 512 tiles
    const int ngrp  = tile >> 9;                     // N = 512  -> 8 groups of 64

    const int arow  = (mtile << 4) + l16;
    const int ncol0 = (ngrp << 6) + l16;

    const unsigned short* __restrict__ xrow = Xb + (size_t)arow * 512 + (half ? 16 : 0);
    const int blo = half ? 16 : 0;

    v8f c[4];
    #pragma unroll
    for (int nt = 0; nt < 4; ++nt) c[nt] = (v8f){0.f,0.f,0.f,0.f,0.f,0.f,0.f,0.f};

    #pragma unroll 4
    for (int k0 = 0; k0 < 512; k0 += 32) {
        Frag16 a;                                    // A: contiguous 0..15 / 16..31
        a.q[0] = *(const uint4*)(xrow + k0);
        a.q[1] = *(const uint4*)(xrow + k0 + 8);
        #pragma unroll
        for (int nt = 0; nt < 4; ++nt) {             // B: contiguous 0..15 / 16..31
            Frag16 bf;
            const unsigned short* bp = Wt + (size_t)(ncol0 + nt * 16) * 512 + k0 + blo;
            bf.q[0] = *(const uint4*)(bp + 0);
            bf.q[1] = *(const uint4*)(bp + 8);
            c[nt] = wmma_bf16(a, bf, c[nt]);
        }
    }

    #pragma unroll
    for (int nt = 0; nt < 4; ++nt) {
        const int   ncol = ncol0 + nt * 16;
        const float bn   = bias[ncol];
        if (mode == 3) {
            float* out = (float*)outp;
            #pragma unroll
            for (int r = 0; r < 8; ++r) {
                int mr = (mtile << 4) + r + (half << 3);     // C layout: row r(+8)
                out[(size_t)mr * 512 + ncol] = c[nt][r] + bn;
            }
        } else {
            unsigned short* out = (unsigned short*)outp;
            const int h = ncol >> 6, d = ncol & 63;
            #pragma unroll
            for (int r = 0; r < 8; ++r) {
                int mr   = (mtile << 4) + r + (half << 3);
                int bidx = mr >> 11;
                int lr   = mr & 2047;
                float val = (c[nt][r] + bn) * scale;
                size_t idx;
                if (mode == 2) idx = ((size_t)((bidx * 8 + h) * 64 + d)) * 2048 + lr;  // V^T
                else           idx = ((size_t)((bidx * 8 + h) * 2048 + lr)) * 64 + d;  // Q/K
                out[idx] = f2bf(val);
            }
        }
    }
}

// ---------------------------------------------------------------------------
// Flash attention: block = 128 q rows of one (b,h); wave = 16 q rows.
// Each 32-KV chunk of K (32x64) and V^T (64x32) is DMA'd into LDS by the TDM,
// double-buffered one chunk ahead; waves read WMMA fragments from LDS.
// ---------------------------------------------------------------------------
__global__ void __launch_bounds__(256) attn_kernel(const unsigned short* __restrict__ Q,
                                                   const unsigned short* __restrict__ K,
                                                   const unsigned short* __restrict__ VT,
                                                   const int* __restrict__ mask,
                                                   unsigned short* __restrict__ XS) {
    const int wave = threadIdx.x >> 5;
    const int lane = threadIdx.x & 31;
    const int l16  = lane & 15;
    const int half = lane >> 4;

    const int id    = blockIdx.x;                 // 512 = 4 * 8 * 16
    const int b     = id >> 7;
    const int h     = (id >> 4) & 7;
    const int qblk  = id & 15;
    const int qrow0 = qblk * 128 + wave * 16;

    const size_t headBase = (size_t)(b * 8 + h) * 2048 * 64;
    const unsigned short* khead = K  + headBase;
    const unsigned short* vhead = VT + (size_t)((b * 8 + h) * 64) * 2048;

    __shared__ __align__(16) unsigned short ldsK[2][32 * 64];  // 2 x 4KB
    __shared__ __align__(16) unsigned short ldsV[2][64 * 32];  // 2 x 4KB
    __shared__ __align__(16) unsigned short ldsP[8][16 * 32];  // per-wave prob tile
    unsigned short* pw = &ldsP[wave][0];

    // Preload Q A-fragments (16 rows x 64 K -> two K=32 fragments), kept in VGPRs.
    Frag16 qa[2];
    {
        const unsigned short* qp = Q + headBase + (size_t)(qrow0 + l16) * 64 + (half ? 16 : 0);
        qa[0].q[0] = *(const uint4*)(qp + 0);
        qa[0].q[1] = *(const uint4*)(qp + 8);
        qa[1].q[0] = *(const uint4*)(qp + 32);
        qa[1].q[1] = *(const uint4*)(qp + 40);
    }

    v8f o[4];
    #pragma unroll
    for (int dt = 0; dt < 4; ++dt) o[dt] = (v8f){0.f,0.f,0.f,0.f,0.f,0.f,0.f,0.f};
    float mrow[8], lrow[8];
    #pragma unroll
    for (int r = 0; r < 8; ++r) { mrow[r] = -1.0e30f; lrow[r] = 0.f; }

    const int* mline = mask + b * 2048;

    // TDM prologue: stage chunk 0
    if (wave == 0) {
        tdm_load_2d(lds_off(&ldsK[0][0]), khead, 64, 32, 64);
        tdm_load_2d(lds_off(&ldsV[0][0]), vhead, 32, 64, 2048);
    }

    for (int it = 0; it < 64; ++it) {
        const int kv0 = it * 32;
        const int cur = it & 1;

        if (wave == 0) {
            if (it + 1 < 64) {   // issue next chunk into other buffer, then wait this one
                tdm_load_2d(lds_off(&ldsK[cur ^ 1][0]),
                            khead + (size_t)(kv0 + 32) * 64, 64, 32, 64);
                tdm_load_2d(lds_off(&ldsV[cur ^ 1][0]),
                            vhead + (kv0 + 32), 32, 64, 2048);
                __builtin_amdgcn_s_wait_tensorcnt((unsigned short)2);
            } else {
                __builtin_amdgcn_s_wait_tensorcnt((unsigned short)0);
            }
        }
        __syncthreads();

        const unsigned short* kc = &ldsK[cur][0];
        const unsigned short* vc = &ldsV[cur][0];

        // ---- scores: two 16(q) x 16(kv) tiles, K-loop over d_k = 2x32 ----
        v8f s0 = (v8f){0.f,0.f,0.f,0.f,0.f,0.f,0.f,0.f};
        v8f s1 = s0;
        #pragma unroll
        for (int ks = 0; ks < 2; ++ks) {
            Frag16 kb0, kb1;
            const unsigned short* kp0 = kc + (size_t)l16 * 64        + ks * 32 + (half ? 16 : 0);
            const unsigned short* kp1 = kc + (size_t)(l16 + 16) * 64 + ks * 32 + (half ? 16 : 0);
            kb0.q[0] = *(const uint4*)(kp0 + 0);
            kb0.q[1] = *(const uint4*)(kp0 + 8);
            kb1.q[0] = *(const uint4*)(kp1 + 0);
            kb1.q[1] = *(const uint4*)(kp1 + 8);
            s0 = wmma_bf16(qa[ks], kb0, s0);
            s1 = wmma_bf16(qa[ks], kb1, s1);
        }

        // ---- mask + online softmax (lane-half == row-half of C layout) ----
        const int mk0 = mline[kv0 + l16];
        const int mk1 = mline[kv0 + 16 + l16];
        float p0[8], p1[8];
        #pragma unroll
        for (int r = 0; r < 8; ++r) {
            float a0 = mk0 ? s0[r] : -3.0e38f;
            float a1 = mk1 ? s1[r] : -3.0e38f;
            float cm = fmaxf(a0, a1);
            cm = fmaxf(cm, __shfl_xor(cm, 1, 32));
            cm = fmaxf(cm, __shfl_xor(cm, 2, 32));
            cm = fmaxf(cm, __shfl_xor(cm, 4, 32));
            cm = fmaxf(cm, __shfl_xor(cm, 8, 32));
            float nm    = fmaxf(mrow[r], cm);
            float alpha = __expf(mrow[r] - nm);
            mrow[r] = nm;
            p0[r] = __expf(a0 - nm);
            p1[r] = __expf(a1 - nm);
            float cs = p0[r] + p1[r];
            cs += __shfl_xor(cs, 1, 32);
            cs += __shfl_xor(cs, 2, 32);
            cs += __shfl_xor(cs, 4, 32);
            cs += __shfl_xor(cs, 8, 32);
            lrow[r] = lrow[r] * alpha + cs;
            o[0][r] *= alpha; o[1][r] *= alpha; o[2][r] *= alpha; o[3][r] *= alpha;
            const int row = r + (half << 3);
            pw[row * 32 + l16]      = f2bf(p0[r]);
            pw[row * 32 + 16 + l16] = f2bf(p1[r]);
        }

        // same-wave LDS RAW: DS ops are in-order, wait them out explicitly
        asm volatile("s_wait_dscnt 0x0" ::: "memory");

        // ---- reload probs in A-fragment layout (16x32 bf16) ----
        Frag16 pf;
        {
            const unsigned short* pr = pw + l16 * 32 + (half ? 16 : 0);
            pf.q[0] = *(const uint4*)(pr + 0);
            pf.q[1] = *(const uint4*)(pr + 8);
        }

        // ---- PV: 4 d-tiles of 16; B-fragments from staged V^T ----
        #pragma unroll
        for (int dt = 0; dt < 4; ++dt) {
            Frag16 vb;
            const unsigned short* vp = vc + (size_t)(dt * 16 + l16) * 32 + (half ? 16 : 0);
            vb.q[0] = *(const uint4*)(vp + 0);
            vb.q[1] = *(const uint4*)(vp + 8);
            o[dt] = wmma_bf16(pf, vb, o[dt]);
        }

        __syncthreads();   // all waves done with buffers before TDM reuses them
    }

    // ---- normalize, write xs as bf16 [B*L, 512] for the output projection ----
    float inv[8];
    #pragma unroll
    for (int r = 0; r < 8; ++r) inv[r] = 1.0f / fmaxf(lrow[r], 1e-30f);
    #pragma unroll
    for (int dt = 0; dt < 4; ++dt) {
        #pragma unroll
        for (int r = 0; r < 8; ++r) {
            int mr = qrow0 + r + (half << 3);
            XS[((size_t)(b * 2048 + mr)) * 512 + h * 64 + dt * 16 + l16] =
                f2bf(o[dt][r] * inv[r]);
        }
    }
}

// ---------------------------------------------------------------------------
// Host-side orchestration (all on `stream`, graph-capture safe).
// ---------------------------------------------------------------------------
extern "C" void kernel_launch(void* const* d_in, const int* in_sizes, int n_in,
                              void* d_out, int out_size, void* d_ws, size_t ws_size,
                              hipStream_t stream) {
    const float* query = (const float*)d_in[0];
    const float* key   = (const float*)d_in[1];
    const float* value = (const float*)d_in[2];
    const int*   mask  = (const int*)  d_in[3];
    const float* Wq    = (const float*)d_in[4];
    const float* bq    = (const float*)d_in[5];
    const float* Wk    = (const float*)d_in[6];
    const float* bk    = (const float*)d_in[7];
    const float* Wv    = (const float*)d_in[8];
    const float* bv    = (const float*)d_in[9];
    const float* Wo    = (const float*)d_in[10];
    const float* bo    = (const float*)d_in[11];

    char* ws = (char*)d_ws;
    const size_t WT_B = (size_t)512 * 512 * 2;            // 512 KB per weight (bf16)
    const size_t XB_B = (size_t)8192 * 512 * 2;           // 8 MB per activation (bf16)
    unsigned short* wtq = (unsigned short*)(ws + 0 * WT_B);
    unsigned short* wtk = (unsigned short*)(ws + 1 * WT_B);
    unsigned short* wtv = (unsigned short*)(ws + 2 * WT_B);
    unsigned short* wto = (unsigned short*)(ws + 3 * WT_B);
    char* act = ws + 4 * WT_B;
    unsigned short* xq = (unsigned short*)(act + 0 * XB_B);
    unsigned short* xk = (unsigned short*)(act + 1 * XB_B);
    unsigned short* xv = (unsigned short*)(act + 2 * XB_B);
    unsigned short* qh = (unsigned short*)(act + 3 * XB_B);
    unsigned short* kh = (unsigned short*)(act + 4 * XB_B);
    unsigned short* vt = (unsigned short*)(act + 5 * XB_B);
    unsigned short* xs = (unsigned short*)(act + 6 * XB_B);

    // 1) weights -> bf16 transposed; activations -> bf16
    wt_conv<<<1024, 256, 0, stream>>>(Wq, wtq);
    wt_conv<<<1024, 256, 0, stream>>>(Wk, wtk);
    wt_conv<<<1024, 256, 0, stream>>>(Wv, wtv);
    wt_conv<<<1024, 256, 0, stream>>>(Wo, wto);
    x_conv<<<2048, 256, 0, stream>>>(query, xq);
    x_conv<<<2048, 256, 0, stream>>>(key,   xk);
    x_conv<<<2048, 256, 0, stream>>>(value, xv);

    // 2) projections (score scale 1/sqrt(64)=0.125 folded into Q)
    gemm_proj<<<512, 256, 0, stream>>>(xq, wtq, bq, qh, 0, 0.125f);
    gemm_proj<<<512, 256, 0, stream>>>(xk, wtk, bk, kh, 1, 1.0f);
    gemm_proj<<<512, 256, 0, stream>>>(xv, wtv, bv, vt, 2, 1.0f);

    // 3) flash attention (TDM-staged K/V)
    attn_kernel<<<512, 256, 0, stream>>>(qh, kh, vt, mask, xs);

    // 4) output projection -> d_out (fp32)
    gemm_proj<<<512, 256, 0, stream>>>(xs, wto, bo, (float*)d_out, 3, 1.0f);
}
